// GATLayer_15882789061167
// MI455X (gfx1250) — compile-verified, matrix operand order checked by reading, not run on hardware
//
#include <hip/hip_runtime.h>
#include <math.h>

#define N_NODES 4096
#define IN_DIM  256
#define OUT_DIM 256
#define HEADS   4
#define KC      272      // 256 value cols + 4 denominator cols, padded to 16
#define LN_EPS  1e-5f

#define KSPLIT   8       // K-ranges of 512 -> 256 workgroups in the main GEMM
#define MBLK     128     // rows per workgroup (8 waves x 16)
#define NTILES   17      // 272/16
#define BS_STRIDE 72     // LDS row stride in halfs (144B = 36 banks, conflict-free)

typedef __attribute__((ext_vector_type(16))) _Float16 v16h;
typedef __attribute__((ext_vector_type(8)))  _Float16 h8;
typedef __attribute__((ext_vector_type(8)))  float    v8f;
typedef __attribute__((ext_vector_type(4)))  float    f4;
typedef __attribute__((ext_vector_type(4)))  int      i4;

// async-to-LDS builtin takes (v4i AS1*, v4i AS3*, imm offset, imm cpol)
typedef __attribute__((address_space(1))) i4 gbl_i4;
typedef __attribute__((address_space(3))) i4 lds_i4;

// ---- WMMA tile helpers -----------------------------------------------------
// 16-bit A-matrix 16x32 layout (ISA 7.12.2): lanes 0-15 hold row M=lane,
// K={0..7} in elems 0..7 and K={16..23} in elems 8..15; lanes 16-31 same rows,
// K shifted +8/+24.  `p` is lane-resolved: base + row*ld + k0 + koff.

static __device__ inline v16h wmma_load_f32(const float* p) {
  f4 a0 = *(const f4*)(p);
  f4 a1 = *(const f4*)(p + 4);
  f4 b0 = *(const f4*)(p + 16);
  f4 b1 = *(const f4*)(p + 20);
  v16h r;
#pragma unroll
  for (int t = 0; t < 4; ++t) {
    r[t]      = (_Float16)a0[t];
    r[4 + t]  = (_Float16)a1[t];
    r[8 + t]  = (_Float16)b0[t];
    r[12 + t] = (_Float16)b1[t];
  }
  return r;
}

// adj values are exactly 0/1: pack two ints into a packed-f16 dword with one
// multiply: (a0 + (a1<<16)) * 0x3C00 -> {f16(a0), f16(a1)}.
static __device__ inline unsigned int pk01(int lo, int hi) {
  return (unsigned int)(lo + (hi << 16)) * 0x3C00u;
}

static __device__ inline v16h wmma_load_adj(const int* p) {
  i4 a0 = *(const i4*)(p);
  i4 a1 = *(const i4*)(p + 4);
  i4 b0 = *(const i4*)(p + 16);
  i4 b1 = *(const i4*)(p + 20);
  union { v16h h; unsigned int u[8]; } t;
  t.u[0] = pk01(a0[0], a0[1]); t.u[1] = pk01(a0[2], a0[3]);
  t.u[2] = pk01(a1[0], a1[1]); t.u[3] = pk01(a1[2], a1[3]);
  t.u[4] = pk01(b0[0], b0[1]); t.u[5] = pk01(b0[2], b0[3]);
  t.u[6] = pk01(b1[0], b1[1]); t.u[7] = pk01(b1[2], b1[3]);
  return t.h;
}

static __device__ inline v16h wmma_load_f16(const _Float16* p) {
  union { v16h v; h8 h[2]; } t;
  t.h[0] = *(const h8*)(p);
  t.h[1] = *(const h8*)(p + 16);
  return t.v;
}

// C/D 16x16 f32 layout: elem v -> (M = v + 8*(lane>=16), N = lane&15)
static __device__ inline void wmma_store(float* q, int ld, v8f d) {
#pragma unroll
  for (int v = 0; v < 8; ++v) q[v * ld] = d[v];
}

// ---- Kernel 0a: init per-head max to -inf ----------------------------------
__global__ void k_init(float* __restrict__ Mh) {
  if (threadIdx.x < HEADS) Mh[threadIdx.x] = __int_as_float(0xff800000);
}

// ---- Kernel 0b: zero the K-split accumulator O -----------------------------
__global__ void __launch_bounds__(256) k_zero(float* __restrict__ O) {
  O[blockIdx.x * 256 + threadIdx.x] = 0.f;   // grid sized exactly N_NODES*KC/256
}

// ---- Kernel 1: hx = x @ W^T  (WMMA, f16 in / f32 acc) ----------------------
__global__ void __launch_bounds__(256) k_hx(const float* __restrict__ x,
                                            const float* __restrict__ W,
                                            float* __restrict__ hx) {
  const int lane = threadIdx.x & 31;
  const int wid  = (blockIdx.x * blockDim.x + threadIdx.x) >> 5;  // 0..4095
  const int mt = wid >> 4;
  const int nt = wid & 15;
  const int r    = lane & 15;
  const int koff = (lane >> 4) << 3;
  const float* arow = x + (mt * 16 + r) * IN_DIM + koff;
  const float* brow = W + (nt * 16 + r) * IN_DIM + koff;   // rows of W = cols of B
  v8f acc = {};
#pragma unroll
  for (int k0 = 0; k0 < IN_DIM; k0 += 32) {
    v16h a = wmma_load_f32(arow + k0);
    v16h b = wmma_load_f32(brow + k0);
    acc = __builtin_amdgcn_wmma_f32_16x16x32_f16(false, a, false, b,
                                                 (short)0, acc, false, false);
  }
  float* q = hx + (mt * 16 + ((lane >> 4) << 3)) * OUT_DIM + nt * 16 + r;
  wmma_store(q, OUT_DIM, acc);
}

// ---- Kernel 2: dst[j,h] = <h[j,h,:], a_dst[h,:]>, per-head global max ------
__global__ void __launch_bounds__(256) k_dst(const float* __restrict__ hx,
                                             const float* __restrict__ a_dst,
                                             float* __restrict__ dstNH,
                                             float* __restrict__ Mh) {
  __shared__ float sh[256];
  const int i = blockIdx.x;
  const int t = threadIdx.x;
  sh[t] = hx[i * OUT_DIM + t] * a_dst[t];
  __syncthreads();
#pragma unroll
  for (int off = 32; off >= 1; off >>= 1) {
    if ((t & 63) < off) sh[t] += sh[t + off];
    __syncthreads();
  }
  if ((t & 63) == 0) {
    const int h = t >> 6;
    const float dv = sh[t];
    dstNH[i * HEADS + h] = dv;
    if (dv >= 0.f) atomicMax((int*)(Mh + h), __float_as_int(dv));
    else           atomicMin((unsigned int*)(Mh + h), __float_as_uint(dv));
  }
}

// ---- Kernel 3: Gt[c,j]: c<256 -> e[j,h(c)]*hx[j,c]; c in [256,260) -> e ----
__global__ void __launch_bounds__(256) k_gt(const float* __restrict__ hx,
                                            const float* __restrict__ dstNH,
                                            const float* __restrict__ Mh,
                                            _Float16* __restrict__ Gt) {
  const int c = blockIdx.y;
  const int j = blockIdx.x * blockDim.x + threadIdx.x;
  float val = 0.f;
  if (c < OUT_DIM) {
    const int h = c >> 6;
    val = __expf(dstNH[j * HEADS + h] - Mh[h]) * hx[j * OUT_DIM + c];
  } else if (c < OUT_DIM + HEADS) {
    const int h = c - OUT_DIM;
    val = __expf(dstNH[j * HEADS + h] - Mh[h]);
  }
  Gt[c * N_NODES + j] = (_Float16)val;
}

// ---- Kernel 4: O += adj_f16 @ G  (block-tiled WMMA GEMM, K-split) ----------
// Grid: 256 WGs = 32 row-blocks x 8 K-splits, 256 threads (8 waves).
// Wave w owns rows mb*128 + w*16, accumulates all 17 N-tiles in registers.
// B K-slice (272 x 64 f16) staged in LDS via GLOBAL_LOAD_ASYNC_TO_LDS_B128
// (s_wait_asynccnt + barrier) and shared by all 8 waves; A tiles converted
// once per stage and reused across all 17 N-tiles.
__global__ void __launch_bounds__(256) k_attn(const int* __restrict__ adj,
                                              const _Float16* __restrict__ Gt,
                                              float* __restrict__ O) {
  __shared__ __attribute__((aligned(16))) _Float16 Bs[KC * BS_STRIDE];

  const int tid  = threadIdx.x;
  const int lane = tid & 31;
  const int w    = tid >> 5;                 // wave 0..7
  const int mb   = blockIdx.x >> 3;          // row block 0..31
  const int kb   = blockIdx.x & 7;           // K split 0..7
  const int r    = lane & 15;
  const int koff = (lane >> 4) << 3;
  const int Kbase = kb * (N_NODES / KSPLIT); // 512-wide K range

  const int* arow = adj + (mb * MBLK + w * 16 + r) * N_NODES + Kbase + koff;

  const v8f vzero = {};
  v8f acc[NTILES];
#pragma unroll
  for (int nt = 0; nt < NTILES; ++nt) acc[nt] = vzero;

  for (int s = 0; s < (N_NODES / KSPLIT) / 64; ++s) {   // 8 stages of K=64
    const int k0 = s * 64;
    __syncthreads();                         // previous stage done reading LDS
    // -- stage Gt[0..271][Kbase+k0 .. +63] into LDS (2176 b128 chunks) ------
    for (int ch = tid; ch < KC * 8; ch += 256) {
      const int c    = ch >> 3;
      const int part = ch & 7;
      const _Float16* gsrc = Gt + c * N_NODES + Kbase + k0 + part * 8;
      _Float16*       ldst = Bs + c * BS_STRIDE + part * 8;
#if defined(__gfx1250__) && __has_builtin(__builtin_amdgcn_global_load_async_to_lds_b128)
      __builtin_amdgcn_global_load_async_to_lds_b128(
          (gbl_i4*)(uintptr_t)gsrc,
          (lds_i4*)(unsigned int)(uintptr_t)ldst, 0, 0);
#else
      *(h8*)ldst = *(const h8*)gsrc;
#endif
    }
#if defined(__gfx1250__) && __has_builtin(__builtin_amdgcn_global_load_async_to_lds_b128)
    __builtin_amdgcn_s_wait_asynccnt(0);
#endif
    __syncthreads();                         // staged slice visible to all waves

    // -- A tiles for this wave's 16 rows, K=k0..k0+63 (kept in VGPRs) -------
    v16h a0 = wmma_load_adj(arow + k0);
    v16h a1 = wmma_load_adj(arow + k0 + 32);

    // -- sweep all 17 N tiles, 2 WMMAs each ---------------------------------
#pragma unroll
    for (int nt = 0; nt < NTILES; ++nt) {
      const _Float16* bp = Bs + (nt * 16 + r) * BS_STRIDE + koff;
      v16h b0 = wmma_load_f16(bp);
      acc[nt] = __builtin_amdgcn_wmma_f32_16x16x32_f16(false, a0, false, b0,
                                                       (short)0, acc[nt], false, false);
      v16h b1 = wmma_load_f16(bp + 32);
      acc[nt] = __builtin_amdgcn_wmma_f32_16x16x32_f16(false, a1, false, b1,
                                                       (short)0, acc[nt], false, false);
    }
  }

  // -- combine K-split partials with relaxed device-scope f32 atomics -------
  const int row0 = mb * MBLK + w * 16 + ((lane >> 4) << 3);
#pragma unroll
  for (int nt = 0; nt < NTILES; ++nt) {
    float* q = O + row0 * KC + nt * 16 + r;
#pragma unroll
    for (int v = 0; v < 8; ++v)
      __hip_atomic_fetch_add(q + v * KC, acc[nt][v],
                             __ATOMIC_RELAXED, __HIP_MEMORY_SCOPE_AGENT);
  }
}

// ---- Kernel 5: out = LayerNorm(O[:, :256]/S + hx) --------------------------
__global__ void __launch_bounds__(256) k_final(const float* __restrict__ O,
                                               const float* __restrict__ hx,
                                               const float* __restrict__ ln_w,
                                               const float* __restrict__ ln_b,
                                               float* __restrict__ out) {
  __shared__ float red[256];
  const int i = blockIdx.x;
  const int t = threadIdx.x;
  const int h = t >> 6;
  const float S   = O[i * KC + OUT_DIM + h];
  const float inv = (S > 0.f) ? 1.0f / S : 0.f;
  const float y   = O[i * KC + t] * inv + hx[i * OUT_DIM + t];
  red[t] = y;
  __syncthreads();
#pragma unroll
  for (int off = 128; off >= 1; off >>= 1) {
    if (t < off) red[t] += red[t + off];
    __syncthreads();
  }
  const float mu = red[0] * (1.0f / OUT_DIM);
  __syncthreads();
  const float d = y - mu;
  red[t] = d * d;
  __syncthreads();
#pragma unroll
  for (int off = 128; off >= 1; off >>= 1) {
    if (t < off) red[t] += red[t + off];
    __syncthreads();
  }
  const float var = red[0] * (1.0f / OUT_DIM);
  out[i * OUT_DIM + t] = d * rsqrtf(var + LN_EPS) * ln_w[t] + ln_b[t];
}

// ---- launch ----------------------------------------------------------------
extern "C" void kernel_launch(void* const* d_in, const int* in_sizes, int n_in,
                              void* d_out, int out_size, void* d_ws, size_t ws_size,
                              hipStream_t stream) {
  (void)in_sizes; (void)n_in; (void)out_size; (void)ws_size;
  const float* x     = (const float*)d_in[0];
  const int*   adj   = (const int*)d_in[1];
  const float* W     = (const float*)d_in[2];
  /* d_in[3] = a_src: cancels in the softmax, unused */
  const float* a_dst = (const float*)d_in[4];
  const float* ln_w  = (const float*)d_in[5];
  const float* ln_b  = (const float*)d_in[6];
  float* out = (float*)d_out;

  char* ws = (char*)d_ws;
  float*    hx    = (float*)(ws);                       // 4096*256*4  = 4 MB
  float*    O     = (float*)(ws + 4194304);             // 4096*272*4  = 4.25 MB
  _Float16* Gt    = (_Float16*)(ws + 8650752);          // 272*4096*2  = 2.125 MB
  float*    dstNH = (float*)(ws + 10878976);            // 4096*4*4
  float*    Mh    = (float*)(ws + 10944512);            // 4 floats

  k_init <<<1, 32, 0, stream>>>(Mh);
  k_hx   <<<512, 256, 0, stream>>>(x, W, hx);
  k_dst  <<<N_NODES, 256, 0, stream>>>(hx, a_dst, dstNH, Mh);
  k_gt   <<<dim3(N_NODES / 256, KC), 256, 0, stream>>>(hx, dstNH, Mh, Gt);
  k_zero <<<(N_NODES * KC) / 256, 256, 0, stream>>>(O);
  k_attn <<<(N_NODES / MBLK) * KSPLIT, 256, 0, stream>>>(adj, Gt, O);
  k_final<<<N_NODES, 256, 0, stream>>>(O, hx, ln_w, ln_b, out);
}
